// CRF_27590869910188
// MI455X (gfx1250) — matrix operand order repaired and checked by reading, not compiled, not run on hardware
//
#include <hip/hip_runtime.h>
#include <hip/hip_bf16.h>

#define PN   4096
#define HH   64
#define WW   64
#define NB   2
#define CC   21
#define CP   32      // channels padded for WMMA N-dim
#define KS   71      // spatial kernel size
#define KR   35      // spatial kernel radius
#define NITER 5

typedef __attribute__((ext_vector_type(16))) __bf16 v16bf;
typedef __attribute__((ext_vector_type(8)))  __bf16 v8bf;
typedef __attribute__((ext_vector_type(8)))  float  v8f;

static __device__ __forceinline__ unsigned short f2bf(float f) {
    unsigned int u = __float_as_uint(f);
    u += 0x7FFFu + ((u >> 16) & 1u);          // round-to-nearest-even
    return (unsigned short)(u >> 16);
}

// ---- bilateral features: [y/70, x/70, r/12, g/12, b/12] per pixel ----------
__global__ void k_feats(const float* __restrict__ ref, const float* __restrict__ kstd,
                        float* __restrict__ feats) {
    int i = blockIdx.x * blockDim.x + threadIdx.x;   // n*PN + p
    if (i >= NB * PN) return;
    int n = i / PN, p = i % PN;
    const float* rp = ref + (size_t)n * 3 * PN + p;
    float* f = feats + (size_t)i * 5;
    f[0] = (float)(p / WW) / kstd[0];
    f[1] = (float)(p % WW) / kstd[1];
    f[2] = rp[0]        / kstd[2];
    f[3] = rp[PN]       / kstd[3];
    f[4] = rp[2 * PN]   / kstd[4];
}

// ---- dense Gaussian kernel K[p,q] = exp(-0.5*||f_p-f_q||^2), stored bf16 ---
__global__ void k_buildK(const float* __restrict__ feats, unsigned short* __restrict__ Kb) {
    __shared__ float fp[16][5], fq[16][5];
    int n = blockIdx.z;
    int p0 = blockIdx.y * 16, q0 = blockIdx.x * 16;
    int t = threadIdx.y * 16 + threadIdx.x;
    if (t < 80)       fp[t / 5][t % 5] = feats[((size_t)n * PN + p0 + t / 5) * 5 + t % 5];
    else if (t < 160) { int u = t - 80; fq[u / 5][u % 5] = feats[((size_t)n * PN + q0 + u / 5) * 5 + u % 5]; }
    __syncthreads();
    int pr = threadIdx.y, qc = threadIdx.x;
    float d2 = 0.f;
#pragma unroll
    for (int d = 0; d < 5; ++d) { float df = fp[pr][d] - fq[qc][d]; d2 += df * df; }
    Kb[((size_t)n * PN + p0 + pr) * PN + q0 + qc] = f2bf(__expf(-0.5f * d2));
}

// ---- U = log(clip(unary)), q0 = softmax(U), qT = bf16 transpose (padded) ---
__global__ void k_init(const float* __restrict__ unary, float* __restrict__ U,
                       float* __restrict__ q, unsigned short* __restrict__ qT) {
    int i = blockIdx.x * blockDim.x + threadIdx.x;
    if (i >= NB * PN) return;
    int n = i / PN, p = i % PN;
    float u[CC], m = -1e30f;
#pragma unroll
    for (int c = 0; c < CC; ++c) {
        float v = unary[((size_t)n * CC + c) * PN + p];
        v = __logf(fminf(fmaxf(v, 1e-5f), 1.0f));
        u[c] = v;
        U[((size_t)n * CC + c) * PN + p] = v;
        m = fmaxf(m, v);
    }
    float s = 0.f;
#pragma unroll
    for (int c = 0; c < CC; ++c) { u[c] = __expf(u[c] - m); s += u[c]; }
    float inv = 1.0f / s;
#pragma unroll
    for (int c = 0; c < CC; ++c) {
        float qv = u[c] * inv;
        q[((size_t)n * CC + c) * PN + p]  = qv;
        qT[((size_t)n * CP + c) * PN + p] = f2bf(qv);
    }
#pragma unroll
    for (int c = CC; c < CP; ++c) qT[((size_t)n * CP + c) * PN + p] = 0;   // zero pad channels
}

// ---- recover exact normalized 1-D separable factor from gk center row ------
static __device__ __forceinline__ void load_g1(const float* __restrict__ gk,
                                               float* s_g, float* s_inv) {
    int t = threadIdx.x;
    if (t < KS) s_g[t] = gk[KR * KS + t];        // channel 0, center row
    __syncthreads();
    if (t == 0) { float s = 0.f; for (int i = 0; i < KS; ++i) s += s_g[i]; *s_inv = 1.0f / s; }
    __syncthreads();
    if (t < KS) s_g[t] *= *s_inv;
    __syncthreads();
}

__global__ void k_conv_h(const float* __restrict__ gk, const float* __restrict__ q,
                         float* __restrict__ tmp) {
    __shared__ float s_g[KS]; __shared__ float s_inv;
    load_g1(gk, s_g, &s_inv);
    int i = blockIdx.x * blockDim.x + threadIdx.x;
    if (i >= NB * CC * PN) return;
    int w = i % WW; int base = i - w;
    int lo = max(0, w - KR), hi = min(WW - 1, w + KR);
    float acc = 0.f;
    for (int x = lo; x <= hi; ++x) acc += s_g[x - w + KR] * q[base + x];
    tmp[i] = acc;
}

__global__ void k_conv_v(const float* __restrict__ gk, const float* __restrict__ tmp,
                         float* __restrict__ tmp2) {
    __shared__ float s_g[KS]; __shared__ float s_inv;
    load_g1(gk, s_g, &s_inv);
    int i = blockIdx.x * blockDim.x + threadIdx.x;
    if (i >= NB * CC * PN) return;
    int w = i % WW; int h = (i / WW) % HH; int base = i - h * WW;
    int lo = max(0, h - KR), hi = min(HH - 1, h + KR);
    float acc = 0.f;
    for (int y = lo; y <= hi; ++y) acc += s_g[y - h + KR] * tmp[base + y * WW];
    tmp2[i] = acc;
}

// ---- qbf[p,c] = sum_q K[p,q] * qT[c,q]  via v_wmma_f32_16x16x32_bf16 -------
__global__ void __launch_bounds__(256) k_gemm(const unsigned short* __restrict__ Kb,
                                              const unsigned short* __restrict__ qTm,
                                              float* __restrict__ qbf) {
    int wave  = threadIdx.x >> 5;
    int lane  = threadIdx.x & 31;
    int gid   = blockIdx.x * 8 + wave;       // 0..511 waves
    int n     = gid >> 8;                    // batch
    int mbase = (gid & 255) * 16;            // 16-row M tile
    int lhalf = lane >> 4;                   // lane group 0/1
    int lmod  = lane & 15;

    // A: K rows, bf16, k = 16*lhalf + s (contiguous), same order as B
    const __bf16* Arow = (const __bf16*)Kb + ((size_t)n * PN + mbase + lmod) * PN;
    const int aoff0 = lhalf * 16, aoff1 = lhalf * 16 + 8;
    // B: qT rows (channel-major), k = 16*lhalf + s (contiguous), same order as A
    const __bf16* B0 = (const __bf16*)qTm + ((size_t)n * CP + lmod) * PN + lhalf * 16;
    const __bf16* B1 = B0 + (size_t)16 * PN;

    v8f acc0 = {}; v8f acc1 = {};
    for (int k = 0; k < PN; k += 32) {
        __builtin_prefetch(Arow + k + 128, 0, 1);      // global_prefetch_b8 on streamed K
        v8bf alo = *(const v8bf*)(Arow + k + aoff0);
        v8bf ahi = *(const v8bf*)(Arow + k + aoff1);
        v16bf a  = __builtin_shufflevector(alo, ahi, 0,1,2,3,4,5,6,7,8,9,10,11,12,13,14,15);
        v16bf b0 = *(const v16bf*)(B0 + k);
        v16bf b1 = *(const v16bf*)(B1 + k);
        acc0 = __builtin_amdgcn_wmma_f32_16x16x32_bf16(false, a, false, b0, (short)0, acc0, false, false);
        acc1 = __builtin_amdgcn_wmma_f32_16x16x32_bf16(false, a, false, b1, (short)0, acc1, false, false);
    }
    float* out = qbf + ((size_t)n * PN + mbase) * CP;
#pragma unroll
    for (int r = 0; r < 8; ++r) {
        int row = r + 8 * lhalf;                       // C/D layout: VGPR r, lane half
        out[row * CP + lmod]      = acc0[r];
        out[row * CP + 16 + lmod] = acc1[r];
    }
}

// ---- q_hat = U + 4*qbf + 2*qsf ; q = softmax_c(q_hat) ----------------------
__global__ void k_combine(const float* __restrict__ U, const float* __restrict__ qbf,
                          const float* __restrict__ qsf, float* __restrict__ qout,
                          unsigned short* __restrict__ qT) {
    int i = blockIdx.x * blockDim.x + threadIdx.x;
    if (i >= NB * PN) return;
    int n = i / PN, p = i % PN;
    float hv[CC], m = -1e30f;
#pragma unroll
    for (int c = 0; c < CC; ++c) {
        float h = U[((size_t)n * CC + c) * PN + p]
                + 4.0f * qbf[((size_t)n * PN + p) * CP + c]
                + 2.0f * qsf[((size_t)n * CC + c) * PN + p];
        hv[c] = h; m = fmaxf(m, h);
    }
    float s = 0.f;
#pragma unroll
    for (int c = 0; c < CC; ++c) { hv[c] = __expf(hv[c] - m); s += hv[c]; }
    float inv = 1.0f / s;
#pragma unroll
    for (int c = 0; c < CC; ++c) {
        float qv = hv[c] * inv;
        qout[((size_t)n * CC + c) * PN + p] = qv;
        qT[((size_t)n * CP + c) * PN + p]   = f2bf(qv);
    }
}

extern "C" void kernel_launch(void* const* d_in, const int* in_sizes, int n_in,
                              void* d_out, int out_size, void* d_ws, size_t ws_size,
                              hipStream_t stream) {
    (void)in_sizes; (void)n_in; (void)out_size; (void)ws_size;
    const float* unary = (const float*)d_in[0];
    const float* ref   = (const float*)d_in[1];
    const float* gk    = (const float*)d_in[2];
    const float* kstd  = (const float*)d_in[3];

    char* ws = (char*)d_ws;
    size_t off = 0;
    auto take = [&](size_t bytes) -> char* {
        char* pp = ws + off;
        off = (off + bytes + 255) & ~(size_t)255;
        return pp;
    };
    unsigned short* Kb   = (unsigned short*)take((size_t)NB * PN * PN * 2);  // 67 MB, L2-resident
    float*          feats= (float*)take((size_t)NB * PN * 5 * 4);
    float*          U    = (float*)take((size_t)NB * CC * PN * 4);
    float*          q    = (float*)take((size_t)NB * CC * PN * 4);
    float*          tmp  = (float*)take((size_t)NB * CC * PN * 4);
    float*          tmp2 = (float*)take((size_t)NB * CC * PN * 4);
    float*          qbf  = (float*)take((size_t)NB * PN * CP * 4);
    unsigned short* qT   = (unsigned short*)take((size_t)NB * CP * PN * 2);

    k_feats <<<(NB * PN + 255) / 256, 256, 0, stream>>>(ref, kstd, feats);
    k_buildK<<<dim3(PN / 16, PN / 16, NB), dim3(16, 16), 0, stream>>>(feats, Kb);
    k_init  <<<(NB * PN + 255) / 256, 256, 0, stream>>>(unary, U, q, qT);

    const int npix = NB * CC * PN;
    for (int it = 0; it < NITER; ++it) {
        k_conv_h<<<(npix + 255) / 256, 256, 0, stream>>>(gk, q, tmp);
        k_conv_v<<<(npix + 255) / 256, 256, 0, stream>>>(gk, tmp, tmp2);
        k_gemm  <<<64, 256, 0, stream>>>(Kb, qT, qbf);
        float* qdst = (it == NITER - 1) ? (float*)d_out : q;
        k_combine<<<(NB * PN + 255) / 256, 256, 0, stream>>>(U, qbf, tmp2, qdst, qT);
    }
}
